// MultiLatentAttention_48206712930695
// MI455X (gfx1250) — matrix operand order repaired and checked by reading, compiled, not run on hardware
//
#include <hip/hip_runtime.h>

#define B_   2
#define S_   2048
#define H_   2048
#define NH_  16
#define LD_  512
#define HD_  128
#define MROWS (B_ * S_)   // 4096

typedef _Float16 v8h  __attribute__((ext_vector_type(8)));
typedef _Float16 v16h __attribute__((ext_vector_type(16)));
typedef float    v8f  __attribute__((ext_vector_type(8)));
typedef unsigned int u32x4 __attribute__((ext_vector_type(4)));
typedef int          i32x4 __attribute__((ext_vector_type(4)));
typedef int          i32x8 __attribute__((ext_vector_type(8)));

__device__ __forceinline__ v16h pack16(v8h lo, v8h hi) {
  return __builtin_shufflevector(lo, hi, 0,1,2,3,4,5,6,7,8,9,10,11,12,13,14,15);
}

// A-fragment (16x32 f16), ISA 7.12.2: lane&15 = M row, lane>>4 = khalf.
__device__ __forceinline__ v16h ldA(const _Float16* rowp, int kb, int kh) {
  v8h lo = *(const v8h*)(rowp + kb + kh * 8);
  v8h hi = *(const v8h*)(rowp + kb + 16 + kh * 8);
  return pack16(lo, hi);
}

// B-fragment (32x16 f16): lane&15 = N column, K = kb + kh*8 + e (e<8), kb + 16 + kh*8 + (e-8) (e>=8)
__device__ __forceinline__ v16h ldB(const _Float16* rowp, int kb, int kh) {
  v8h lo = *(const v8h*)(rowp + kb + kh * 8);
  v8h hi = *(const v8h*)(rowp + kb + 16 + kh * 8);
  return pack16(lo, hi);
}

__device__ __forceinline__ v8f wmma16(v16h a, v16h b, v8f c) {
  return __builtin_amdgcn_wmma_f32_16x16x32_f16(false, a, false, b, (short)0, c,
                                                false, false);
}

// ------------------------------------------------------------- TDM 2D loader
// Load a tile_d1 x tile_d0 (rows x contiguous f16 elems) tile from global
// (row stride = stride_d0 elems) into packed LDS at lds_dst.
// D# packing per CDNA5 ISA.
__device__ __forceinline__ void tdm_load_2d(const void* gsrc,
                                            const void* lds_dst,
                                            unsigned tile_d0, unsigned tile_d1,
                                            unsigned stride_d0) {
  unsigned long long ga = (unsigned long long)(uintptr_t)gsrc;
  u32x4 g0;
  g0[0] = 1u;                                   // count=1, user descriptor
  g0[1] = (unsigned)(uintptr_t)lds_dst;         // LDS byte offset (addr[31:0])
  g0[2] = (unsigned)ga;                         // global_addr[31:0]
  g0[3] = (unsigned)(ga >> 32) | (2u << 30);    // global_addr[56:32] | type=2
  i32x8 g1;
  g1[0] = 0x10000;                 // workgroup_mask=0, data_size=1 (2 bytes)
  g1[1] = 0;                       // bar_addr=0, tensor_dim0[15:0]=0
  g1[2] = 0x4000;                  // tensor_dim0=0x40000000, dim1[15:0]=0
  g1[3] = 0x4000 | (int)(tile_d0 << 16);  // tensor_dim1=0x40000000 | tile_dim0
  g1[4] = (int)tile_d1;            // tile_dim1, tile_dim2=0 (2D)
  g1[5] = (int)stride_d0;          // tensor_dim0_stride[31:0]
  g1[6] = 0;
  g1[7] = 0;
  i32x4 gz = {0, 0, 0, 0};
#if defined(__clang_major__) && (__clang_major__ >= 23)
  i32x8 gz8 = {0, 0, 0, 0, 0, 0, 0, 0};
  __builtin_amdgcn_tensor_load_to_lds(g0, g1, gz, gz, gz8, 0);
#else
  __builtin_amdgcn_tensor_load_to_lds(g0, g1, gz, gz, 0);
#endif
}

// ---------------------------------------------------------------- conversions
__global__ __launch_bounds__(256) void cvt_f32_f16(const float* __restrict__ in,
                                                   _Float16* __restrict__ out,
                                                   int n) {
  int i = blockIdx.x * 256 + threadIdx.x;
  if (i < n) out[i] = (_Float16)in[i];
}

// in: [R][C] f32  ->  out: [C][R] f16   (weights: [K][N] -> [N][K])
__global__ __launch_bounds__(256) void transpose_f32_f16(
    const float* __restrict__ in, _Float16* __restrict__ out, int R, int C) {
  int i = blockIdx.x * 256 + threadIdx.x;
  if (i < R * C) {
    int r = i / C, c = i % C;
    out[(size_t)c * R + r] = (_Float16)in[i];
  }
}

// ---------------------------------------------------------------- WMMA GEMM
// C[M,N] = A[M,K](f16) @ Bt[N,K](f16)^T  (+ bias[N])
// Block tile 128x128x64, 256 threads = 8 waves; wave tile 32x64 (2x4 WMMAs).
// Tiles are staged by the Tensor Data Mover (wave 0 issues, TENSORcnt-waited),
// double-buffered so the DMA of tile t+1 overlaps compute of tile t.
#define BM 128
#define BN 128
#define BKT 64

template <bool HALF_OUT, bool HAS_BIAS>
__global__ __launch_bounds__(256) void gemm_wmma(
    const _Float16* __restrict__ A, const _Float16* __restrict__ Bt,
    const float* __restrict__ bias, void* __restrict__ Cout,
    int M, int N, int K) {
  __shared__ _Float16 As[2][BM][BKT];   // 2 x 16 KB
  __shared__ _Float16 Bs[2][BN][BKT];   // 2 x 16 KB

  const int t    = threadIdx.x;
  const int lane = t & 31;
  const int wave = t >> 5;
  const int wm   = wave >> 1;   // 0..3 : 32-row strip
  const int wn   = wave & 1;    // 0..1 : 64-col strip
  const int m0   = blockIdx.y * BM;
  const int n0   = blockIdx.x * BN;
  const int ml   = lane & 15;
  const int kh   = lane >> 4;

  v8f acc[2][4];
#pragma unroll
  for (int i = 0; i < 2; ++i)
#pragma unroll
    for (int j = 0; j < 4; ++j) acc[i][j] = (v8f)0.0f;

  const int nk = K / BKT;
  if (wave == 0) {  // prologue DMA: tile 0 into buffer 0
    tdm_load_2d(&A[(size_t)m0 * K], &As[0][0][0], BKT, BM, (unsigned)K);
    tdm_load_2d(&Bt[(size_t)n0 * K], &Bs[0][0][0], BKT, BM, (unsigned)K);
  }

  for (int kt = 0; kt < nk; ++kt) {
    if (wave == 0) __builtin_amdgcn_s_wait_tensorcnt(0);  // tile kt landed
    __syncthreads();  // publishes tile kt; proves compute kt-1 done (buf reuse)
    if (wave == 0 && kt + 1 < nk) {  // DMA tile kt+1 into the other buffer
      const int k0 = (kt + 1) * BKT;
      const int nb = (kt + 1) & 1;
      tdm_load_2d(&A[(size_t)m0 * K + k0], &As[nb][0][0], BKT, BM, (unsigned)K);
      tdm_load_2d(&Bt[(size_t)n0 * K + k0], &Bs[nb][0][0], BKT, BM, (unsigned)K);
    }
    const int buf = kt & 1;

#pragma unroll
    for (int ks = 0; ks < 2; ++ks) {
      const int kb = ks * 32;
      v16h af[2], bf[4];
#pragma unroll
      for (int i = 0; i < 2; ++i)
        af[i] = ldA(&As[buf][wm * 32 + i * 16 + ml][0], kb, kh);
#pragma unroll
      for (int j = 0; j < 4; ++j)
        bf[j] = ldB(&Bs[buf][wn * 64 + j * 16 + ml][0], kb, kh);
#pragma unroll
      for (int i = 0; i < 2; ++i)
#pragma unroll
        for (int j = 0; j < 4; ++j) acc[i][j] = wmma16(af[i], bf[j], acc[i][j]);
    }
  }

  // Epilogue: C layout (ISA 7.12.2): VGPR r -> row r + 8*(lane>>4), col lane&15
#pragma unroll
  for (int i = 0; i < 2; ++i)
#pragma unroll
    for (int j = 0; j < 4; ++j) {
      const int col = n0 + wn * 64 + j * 16 + ml;
      float b = HAS_BIAS ? bias[col] : 0.0f;
#pragma unroll
      for (int r = 0; r < 8; ++r) {
        const int row = m0 + wm * 32 + i * 16 + r + 8 * kh;
        float v = acc[i][j][r] + b;
        if (HALF_OUT)
          ((_Float16*)Cout)[(size_t)row * N + col] = (_Float16)v;
        else
          ((float*)Cout)[(size_t)row * N + col] = v;
      }
    }
}

// ------------------------------------------------------- flash MLA attention
// grid = (S/128, NH, B), block = 256 (8 waves). Wave w handles q rows
// [q0 + 16w, q0 + 16w + 16). K tile 64x128 and V^T tile 128x64 shared in LDS.
__global__ __launch_bounds__(256) void mla_attention(
    const _Float16* __restrict__ Qg, const _Float16* __restrict__ Kg,
    const _Float16* __restrict__ Vg, _Float16* __restrict__ Og) {
  __shared__ _Float16 Kt[64][HD_];    // [kv][hd]      16 KB
  __shared__ _Float16 Vt[HD_][64];    // [hd][kv]      16 KB
  __shared__ _Float16 Pb[8][16][64];  // per-wave P    16 KB

  const int t    = threadIdx.x;
  const int lane = t & 31;
  const int wave = t >> 5;
  const int ml   = lane & 15;
  const int kh   = lane >> 4;
  const int h    = blockIdx.y;
  const int b    = blockIdx.z;
  const int q0   = blockIdx.x * 128;
  const int qr   = q0 + wave * 16;          // first q row of this wave
  const float scale = 0.08838834764831845f; // 1/sqrt(128)

  // Q fragments for this wave's 16 rows (K=128 -> 4 chunks of 32), kept live.
  v16h qf[4];
  {
    const _Float16* qbase = Qg + ((size_t)(b * S_ + qr + ml)) * H_ + h * HD_;
#pragma unroll
    for (int ks = 0; ks < 4; ++ks) {
      v8h lo = *(const v8h*)(qbase + ks * 32 + kh * 8);
      v8h hi = *(const v8h*)(qbase + ks * 32 + 16 + kh * 8);
      qf[ks] = pack16(lo, hi);
    }
  }

  v8f o[8];
#pragma unroll
  for (int j = 0; j < 8; ++j) o[j] = (v8f)0.0f;
  float mrun[8], lrun[8];
#pragma unroll
  for (int r = 0; r < 8; ++r) { mrun[r] = -3.0e38f; lrun[r] = 0.0f; }

  const int ntiles = q0 / 64 + 2;  // covers causal span for all 8 waves
  for (int kt = 0; kt < ntiles; ++kt) {
    const int kv0 = kt * 64;
    __syncthreads();
    // Stage K tile (row-major) and V tile (transposed): 1024 chunks each.
#pragma unroll
    for (int i = 0; i < 4; ++i) {
      int c   = t + 256 * i;
      int row = c >> 4;           // kv row 0..63
      int col = (c & 15) * 8;     // hd col
      *(v8h*)&Kt[row][col] =
          *(const v8h*)&Kg[((size_t)(b * S_ + kv0 + row)) * H_ + h * HD_ + col];
      v8h vv =
          *(const v8h*)&Vg[((size_t)(b * S_ + kv0 + row)) * H_ + h * HD_ + col];
#pragma unroll
      for (int e = 0; e < 8; ++e) Vt[col + e][row] = vv[e];
    }
    __syncthreads();

    if (kv0 > qr + 15) continue;  // causal: this wave done (barriers stay uniform)

    // S = Q @ K^T  (16 x 64), contraction over HD=128
    v8f s[4];
#pragma unroll
    for (int f = 0; f < 4; ++f) s[f] = (v8f)0.0f;
#pragma unroll
    for (int ks = 0; ks < 4; ++ks) {
#pragma unroll
      for (int f = 0; f < 4; ++f) {
        v16h bf = ldB(&Kt[f * 16 + ml][0], ks * 32, kh);
        s[f] = wmma16(qf[ks], bf, s[f]);
      }
    }

    // Online softmax per row; rows r+8*kh live across the 16 lanes of a half.
#pragma unroll
    for (int r = 0; r < 8; ++r) {
      const int grow = qr + r + 8 * kh;
      float mx = -3.0e38f;
#pragma unroll
      for (int f = 0; f < 4; ++f) {
        const int gcol = kv0 + f * 16 + ml;
        float x = s[f][r] * scale;
        if (gcol > grow) x = -3.0e38f;
        s[f][r] = x;
        mx = fmaxf(mx, x);
      }
      mx = fmaxf(mx, __shfl_xor(mx, 1, 32));
      mx = fmaxf(mx, __shfl_xor(mx, 2, 32));
      mx = fmaxf(mx, __shfl_xor(mx, 4, 32));
      mx = fmaxf(mx, __shfl_xor(mx, 8, 32));
      const float mn    = fmaxf(mrun[r], mx);
      const float alpha = __expf(mrun[r] - mn);
      mrun[r] = mn;
      float rs = 0.0f;
#pragma unroll
      for (int f = 0; f < 4; ++f) {
        float p = __expf(s[f][r] - mn);
        Pb[wave][r + 8 * kh][f * 16 + ml] = (_Float16)p;
        rs += p;
      }
      rs += __shfl_xor(rs, 1, 32);
      rs += __shfl_xor(rs, 2, 32);
      rs += __shfl_xor(rs, 4, 32);
      rs += __shfl_xor(rs, 8, 32);
      lrun[r] = lrun[r] * alpha + rs;
#pragma unroll
      for (int j = 0; j < 8; ++j) o[j][r] *= alpha;
    }

    // Ensure per-wave LDS stores of P land before re-reading as A-fragments.
    asm volatile("s_wait_dscnt 0x0" ::: "memory");

    // O += P(16x64) @ V(64x128)
#pragma unroll
    for (int ks = 0; ks < 2; ++ks) {
      v16h pf = ldA(&Pb[wave][ml][0], ks * 32, kh);
#pragma unroll
      for (int j = 0; j < 8; ++j) {
        v16h bf = ldB(&Vt[j * 16 + ml][0], ks * 32, kh);
        o[j] = wmma16(pf, bf, o[j]);
      }
    }
  }

  // Normalize and write f16 output at [b, row, h*HD + col]
#pragma unroll
  for (int r = 0; r < 8; ++r) {
    const float inv  = 1.0f / lrun[r];
    const int   grow = qr + r + 8 * kh;
    const size_t base = ((size_t)(b * S_ + grow)) * H_ + h * HD_;
#pragma unroll
    for (int j = 0; j < 8; ++j)
      Og[base + j * 16 + ml] = (_Float16)(o[j][r] * inv);
  }
}

// ---------------------------------------------------------------- launcher
extern "C" void kernel_launch(void* const* d_in, const int* in_sizes, int n_in,
                              void* d_out, int out_size, void* d_ws,
                              size_t ws_size, hipStream_t stream) {
  (void)in_sizes; (void)n_in; (void)out_size; (void)ws_size;

  const float* x      = (const float*)d_in[0];
  const float* wq     = (const float*)d_in[1];
  const float* bq     = (const float*)d_in[2];
  const float* wk_lat = (const float*)d_in[3];
  const float* wv_lat = (const float*)d_in[4];
  const float* wk     = (const float*)d_in[5];
  const float* wv     = (const float*)d_in[6];
  const float* wo     = (const float*)d_in[7];
  const float* bo     = (const float*)d_in[8];
  float* out = (float*)d_out;

  // Workspace carve-up (256-byte aligned).
  char* p = (char*)d_ws;
  auto alloc = [&](size_t elems) {
    _Float16* r = (_Float16*)p;
    p += (elems * sizeof(_Float16) + 255) & ~(size_t)255;
    return r;
  };
  _Float16* xh    = alloc((size_t)MROWS * H_);
  _Float16* wqt   = alloc((size_t)H_ * H_);    // [N=H][K=H]
  _Float16* wklt  = alloc((size_t)LD_ * H_);   // [N=LD][K=H]
  _Float16* wvlt  = alloc((size_t)LD_ * H_);
  _Float16* wkt   = alloc((size_t)H_ * LD_);   // [N=H][K=LD]
  _Float16* wvt   = alloc((size_t)H_ * LD_);
  _Float16* wot   = alloc((size_t)H_ * H_);
  _Float16* qh    = alloc((size_t)MROWS * H_);
  _Float16* klath = alloc((size_t)MROWS * LD_);
  _Float16* vlath = alloc((size_t)MROWS * LD_);
  _Float16* kh_   = alloc((size_t)MROWS * H_);
  _Float16* vh_   = alloc((size_t)MROWS * H_);
  _Float16* attnh = alloc((size_t)MROWS * H_);

  // 1) precision conversion + weight transposition (one-time, bandwidth-cheap)
  {
    int n = MROWS * H_;
    cvt_f32_f16<<<(n + 255) / 256, 256, 0, stream>>>(x, xh, n);
    int nw = H_ * H_;
    transpose_f32_f16<<<(nw + 255) / 256, 256, 0, stream>>>(wq, wqt, H_, H_);
    transpose_f32_f16<<<(nw + 255) / 256, 256, 0, stream>>>(wo, wot, H_, H_);
    int nl = H_ * LD_;
    transpose_f32_f16<<<(nl + 255) / 256, 256, 0, stream>>>(wk_lat, wklt, H_, LD_);
    transpose_f32_f16<<<(nl + 255) / 256, 256, 0, stream>>>(wv_lat, wvlt, H_, LD_);
    transpose_f32_f16<<<(nl + 255) / 256, 256, 0, stream>>>(wk, wkt, LD_, H_);
    transpose_f32_f16<<<(nl + 255) / 256, 256, 0, stream>>>(wv, wvt, LD_, H_);
  }

  // 2) projections (TDM-staged, double-buffered WMMA GEMMs)
  {
    dim3 blk(256);
    dim3 gq(H_ / BN, MROWS / BM);
    gemm_wmma<true, true><<<gq, blk, 0, stream>>>(xh, wqt, bq, qh,
                                                  MROWS, H_, H_);
    dim3 gl(LD_ / BN, MROWS / BM);
    gemm_wmma<true, false><<<gl, blk, 0, stream>>>(xh, wklt, nullptr, klath,
                                                   MROWS, LD_, H_);
    gemm_wmma<true, false><<<gl, blk, 0, stream>>>(xh, wvlt, nullptr, vlath,
                                                   MROWS, LD_, H_);
    dim3 gkv(H_ / BN, MROWS / BM);
    gemm_wmma<true, false><<<gkv, blk, 0, stream>>>(klath, wkt, nullptr, kh_,
                                                    MROWS, H_, LD_);
    gemm_wmma<true, false><<<gkv, blk, 0, stream>>>(vlath, wvt, nullptr, vh_,
                                                    MROWS, H_, LD_);
  }

  // 3) causal flash attention (per head, HD=128)
  {
    dim3 grid(S_ / 128, NH_, B_);
    mla_attention<<<grid, 256, 0, stream>>>(qh, kh_, vh_, attnh);
  }

  // 4) output projection -> f32 result
  {
    dim3 go(H_ / BN, MROWS / BM);
    gemm_wmma<false, true><<<go, 256, 0, stream>>>(attnh, wot, bo, out,
                                                   MROWS, H_, H_);
  }
}
